// CustomLSTM_11982958755994
// MI455X (gfx1250) — matrix operand order repaired and checked by reading, not compiled, not run on hardware
//
#include <hip/hip_runtime.h>
#include <cstdint>
#include <cstddef>

// ---------------------------------------------------------------------------
// Bidirectional LSTM for MI455X (gfx1250), wave32 + bf16 WMMA + TDM.
//  Phase 1: chip-wide GEMM  G[d] = X @ W_ih_d^T  (TDM-staged A, bf16 wmma)
//  Phase 2: 2 independent scans, 32 blocks/dir; W_hh panels TDM-staged in LDS
//           once, h_prev TDM-staged per step; per-step device barrier.
// ---------------------------------------------------------------------------

typedef __bf16 bf16_t;
typedef __attribute__((ext_vector_type(8)))  bf16_t bf16x8;
typedef __attribute__((ext_vector_type(16))) bf16_t bf16x16;
typedef __attribute__((ext_vector_type(8)))  float  f32x8;
typedef unsigned int u32x4 __attribute__((ext_vector_type(4)));
typedef int          i32x4 __attribute__((ext_vector_type(4)));
typedef int          i32x8 __attribute__((ext_vector_type(8)));

#define T_LEN 512
#define B_SZ  64
#define I_SZ  512
#define H_SZ  512
#define G4H   2048                 // 4*H
#define ROWS  (T_LEN * B_SZ)      // 32768
#define MT    (ROWS / 16)         // 2048 m-tiles
#define NT    (G4H / 16)          // 128 n-tiles
#define SCAN_BLOCKS_PER_DIR 32

// ---- TDM (Tensor Data Mover) support --------------------------------------
#if defined(__has_builtin)
#  if __has_builtin(__builtin_amdgcn_tensor_load_to_lds)
#    define HAVE_TDM 1
#  endif
#endif
#ifndef HAVE_TDM
#  define HAVE_TDM 0
#endif

static __device__ __forceinline__ uint32_t lds_off(const void* p) {
  // generic pointer to LDS: low 32 bits are the workgroup-relative offset
  return (uint32_t)(uintptr_t)p;
}

#if HAVE_TDM
// 2D bf16 tile load: global (row-major, stride_d0 elements) -> LDS (packed).
// D# packing per CDNA5 ISA.
static __device__ __forceinline__ void
tdm_load_2d(uint32_t lds_byte_off, const void* gsrc,
            uint32_t tensor_d0, uint32_t tensor_d1,
            uint32_t tile_d0, uint32_t tile_d1, uint32_t stride_d0) {
  const uint64_t ga = (uint64_t)(uintptr_t)gsrc;
  u32x4 g0;
  g0[0] = 1u;                                             // count=1, user mode
  g0[1] = lds_byte_off;                                   // lds_addr [63:32]
  g0[2] = (uint32_t)ga;                                   // global_addr lo
  g0[3] = (uint32_t)((ga >> 32) & 0x1FFFFFFu) | (2u << 30);  // addr hi | type=2
  i32x8 g1;
  g1[0] = (int)(1u << 16);                                // data_size=1 (2B)
  g1[1] = (int)(tensor_d0 << 16);                         // tensor_dim0[15:0]
  g1[2] = (int)((tensor_d0 >> 16) | (tensor_d1 << 16));   // d0 hi | d1 lo
  g1[3] = (int)((tensor_d1 >> 16) | (tile_d0 << 16));     // d1 hi | tile_dim0
  g1[4] = (int)(tile_d1 & 0xFFFFu);                       // tile_dim1, dim2=0
  g1[5] = (int)stride_d0;                                 // dim0_stride lo32
  g1[6] = 0;                                              // stride hi, d1_stride
  g1[7] = 0;
  const i32x4 z4 = {0, 0, 0, 0};
#if __clang_major__ >= 23
  const i32x8 z8 = {0, 0, 0, 0, 0, 0, 0, 0};
  __builtin_amdgcn_tensor_load_to_lds(g0, g1, z4, z4, z8, 0);
#else
  __builtin_amdgcn_tensor_load_to_lds(g0, g1, z4, z4, 0);
#endif
}

static __device__ __forceinline__ void tdm_wait() {
#if __has_builtin(__builtin_amdgcn_s_wait_tensorcnt)
  __builtin_amdgcn_s_wait_tensorcnt(0);
#else
  asm volatile("s_wait_tensorcnt 0x0" ::: "memory");
#endif
}
#endif  // HAVE_TDM

// ---- WMMA fragment helpers (layouts per CDNA5 ISA §7.12.2) ------

// A 16x32 bf16: lane<16 row=lane K{k..k+7, k+16..k+23}; lane>=16 row=lane-16
// K{k+8..k+15, k+24..k+31}. Two 16B contiguous runs.
static __device__ __forceinline__ bf16x16
load_a_frag(const bf16_t* __restrict__ base, int ld, int row0, int k0, int lane) {
  const int m  = row0 + (lane & 15);
  const int kb = k0 + ((lane >> 4) << 3);
  const bf16_t* p = base + (size_t)m * ld + kb;
  bf16x8 lo = *reinterpret_cast<const bf16x8*>(p);
  bf16x8 hi = *reinterpret_cast<const bf16x8*>(p + 16);
  return __builtin_shufflevector(lo, hi, 0,1,2,3,4,5,6,7,8,9,10,11,12,13,14,15);
}

// B 32x16 bf16 with B = W^T (W row-major [N x K]): lane<16 col n0+lane,
// K{k0..k0+7, k0+16..k0+23}; lane>=16 same col, K{k0+8..k0+15, k0+24..k0+31}.
static __device__ __forceinline__ bf16x16
load_b_frag(const bf16_t* __restrict__ w, int ld, int n0, int k0, int lane) {
  const int n  = n0 + (lane & 15);
  const int kb = k0 + ((lane >> 4) << 3);
  const bf16_t* p = w + (size_t)n * ld + kb;
  bf16x8 lo = *reinterpret_cast<const bf16x8*>(p);
  bf16x8 hi = *reinterpret_cast<const bf16x8*>(p + 16);
  return __builtin_shufflevector(lo, hi, 0,1,2,3,4,5,6,7,8,9,10,11,12,13,14,15);
}

static __device__ __forceinline__ f32x8
wmma_bf16(bf16x16 a, bf16x16 b, f32x8 c) {
  return __builtin_amdgcn_wmma_f32_16x16x32_bf16(false, a, false, b, (short)0, c,
                                                 false, false);
}

// ---- Small prep kernels ----------------------------------------------------

__global__ void __launch_bounds__(256)
f32_to_bf16_kernel(const float* __restrict__ in, bf16_t* __restrict__ out, int n) {
  int i = blockIdx.x * 256 + threadIdx.x;
  if (i < n) out[i] = (bf16_t)in[i];
}

__global__ void __launch_bounds__(256)
bias_kernel(const float* __restrict__ bif, const float* __restrict__ bhf,
            const float* __restrict__ bib, const float* __restrict__ bhb,
            float* __restrict__ out) {
  int i = blockIdx.x * 256 + threadIdx.x;       // 0..4095
  if (i < G4H)          out[i] = bif[i] + bhf[i];
  else if (i < 2 * G4H) out[i] = bib[i - G4H] + bhb[i - G4H];
}

__global__ void __launch_bounds__(256)
init_state_kernel(uint32_t* __restrict__ cnt, bf16_t* __restrict__ hbuf) {
  int i = blockIdx.x * 256 + threadIdx.x;       // 131072 threads
  if (i < 2 * T_LEN) cnt[i] = 0u;
  hbuf[i] = (bf16_t)0.0f;                       // 2 dir * 2 buf * 64 * 512
}

// ---- Phase 1: G[d] = X @ W_ih_d^T ------------------------------------------
// grid = 2*512*16; block macro-tile = 64 rows x 128 cols. X block-row (64x512,
// 64KB) staged into LDS by one TDM op; B fragments straight from L2-resident W.
// G stored fragment-tiled: [d][mtile][ntile][lane][8] -> one b128 per tile.

__global__ void __launch_bounds__(256)
input_gemm_kernel(const bf16_t* __restrict__ X,       // [32768 x 512]
                  const bf16_t* __restrict__ Wf,      // [2048 x 512]
                  const bf16_t* __restrict__ Wb,
                  bf16_t* __restrict__ G) {           // [2][MT][NT][32][8]
  __shared__ __attribute__((aligned(16))) bf16_t lds_x[B_SZ * I_SZ];  // 64 KB
  int idx = blockIdx.x;
  const int nb = idx & 15;  idx >>= 4;
  const int mb = idx & 511; idx >>= 9;
  const int d  = idx;
  const bf16_t* __restrict__ W = d ? Wb : Wf;
  const int wave = threadIdx.x >> 5, lane = threadIdx.x & 31;
  const int mBase = mb * 64;
  const int n0 = nb * 128 + wave * 16;

#if HAVE_TDM
  if (wave == 0) {
    tdm_load_2d(lds_off(lds_x), X + (size_t)mBase * I_SZ,
                I_SZ, 64, I_SZ, 64, I_SZ);
    tdm_wait();
  }
#else
  for (int i = threadIdx.x * 8; i < B_SZ * I_SZ; i += 256 * 8)
    *reinterpret_cast<bf16x8*>(lds_x + i) =
        *reinterpret_cast<const bf16x8*>(X + (size_t)mBase * I_SZ + i);
#endif
  __syncthreads();

  f32x8 acc[4] = {};
  for (int k = 0; k < I_SZ; k += 32) {
    if (k + 64 < I_SZ)   // global_prefetch_b8 hint for upcoming W K-slice
      __builtin_prefetch(W + (size_t)(n0 + (lane & 15)) * I_SZ + k + 64, 0, 1);
    bf16x16 bfrag = load_b_frag(W, I_SZ, n0, k, lane);
#pragma unroll
    for (int mt = 0; mt < 4; ++mt) {
      bf16x16 afrag = load_a_frag(lds_x, I_SZ, mt * 16, k, lane);
      acc[mt] = wmma_bf16(afrag, bfrag, acc[mt]);
    }
  }

  bf16_t* __restrict__ Gd = G + (size_t)d * MT * NT * 256;
  const int nt = nb * 8 + wave;
#pragma unroll
  for (int mt = 0; mt < 4; ++mt) {
    bf16x8 pk;
#pragma unroll
    for (int r = 0; r < 8; ++r) pk[r] = (bf16_t)acc[mt][r];
    *reinterpret_cast<bf16x8*>(
        Gd + ((((size_t)(mb * 4 + mt)) * NT + nt) * 32 + lane) * 8) = pk;
  }
}

// ---- Phase 2: recurrent scan ----------------------------------------------
// 64 blocks: dir = bid>>5, blk = bid&31 owns h-cols [blk*16, blk*16+16).
// W_hh panels (4 gates x 16 rows x 512, 64KB) staged by TDM once; h_prev
// (64x512 bf16, 64KB) staged by TDM each step. K-loop = ds_load + wmma only.

__global__ void __launch_bounds__(256)
lstm_scan_kernel(const bf16_t* __restrict__ G,       // [2][MT][NT][32][8]
                 const bf16_t* __restrict__ Whh_f,   // [2048][512] bf16
                 const bf16_t* __restrict__ Whh_b,
                 const float*  __restrict__ biasws,  // [2][2048]
                 bf16_t* __restrict__ hbuf,          // [2 dir][2 buf][64][512]
                 uint32_t* __restrict__ cnt,         // [2][512]
                 float* __restrict__ out) {          // [512][64][1024]
  __shared__ __attribute__((aligned(16))) bf16_t lds_w[64 * H_SZ];   // 64 KB
  __shared__ __attribute__((aligned(16))) bf16_t lds_h[B_SZ * H_SZ]; // 64 KB
  __shared__ float gbuf[4][B_SZ][16];                                // 16 KB
  __shared__ float cbuf[B_SZ * 16];                                  //  4 KB

  const int dir = blockIdx.x >> 5;
  const int blk = blockIdx.x & 31;
  const int tid = threadIdx.x;
  const int wave = tid >> 5, lane = tid & 31;
  const bf16_t* __restrict__ Whh = dir ? Whh_b : Whh_f;

  // Stage this block's 4 W_hh gate panels once (reused for all 512 steps).
#if HAVE_TDM
  if (wave == 0) {
    for (int gg = 0; gg < 4; ++gg)
      tdm_load_2d(lds_off(lds_w) + (uint32_t)gg * 16 * H_SZ * 2,
                  Whh + (size_t)(gg * H_SZ + blk * 16) * H_SZ,
                  H_SZ, 16, H_SZ, 16, H_SZ);
    tdm_wait();
  }
#else
  for (int gg = 0; gg < 4; ++gg)
    for (int i = tid * 8; i < 16 * H_SZ; i += 256 * 8)
      *reinterpret_cast<bf16x8*>(lds_w + gg * 16 * H_SZ + i) =
          *reinterpret_cast<const bf16x8*>(
              Whh + (size_t)(gg * H_SZ + blk * 16) * H_SZ + i);
#endif
#pragma unroll
  for (int q = 0; q < 4; ++q) cbuf[tid + 256 * q] = 0.0f;
  __syncthreads();

  const int g   = wave >> 1;
  const int mtA = (wave & 1) * 2;                 // two mtiles: mtA, mtA+1
  const int n0  = g * H_SZ + blk * 16;            // column in [0,4H)
  const int nt  = g * 32 + blk;                   // ntile in [0,128)
  const int nc  = lane & 15;
  const float bias_n = biasws[dir * G4H + n0 + nc];
  const bf16_t* __restrict__ Gdp = G + (size_t)dir * MT * NT * 256;
  uint32_t* __restrict__ mycnt = cnt + dir * T_LEN;

  for (int s = 0; s < T_LEN; ++s) {
    const int t = dir ? (T_LEN - 1 - s) : s;
    const bf16_t* __restrict__ hin =
        hbuf + (size_t)(dir * 2 + (s & 1)) * B_SZ * H_SZ;
    bf16_t* __restrict__ hout =
        hbuf + (size_t)(dir * 2 + ((s + 1) & 1)) * B_SZ * H_SZ;

    // Kick off h_prev DMA, then fetch this step's x-projection fragments
    // (independent of h) while the DMA is in flight.
#if HAVE_TDM
    if (wave == 0)
      tdm_load_2d(lds_off(lds_h), hin, H_SZ, B_SZ, H_SZ, B_SZ, H_SZ);
#else
    for (int i = tid * 8; i < B_SZ * H_SZ; i += 256 * 8)
      *reinterpret_cast<bf16x8*>(lds_h + i) =
          *reinterpret_cast<const bf16x8*>(hin + i);
#endif
    const bf16x8 gp0 = *reinterpret_cast<const bf16x8*>(
        Gdp + ((((size_t)(t * 4 + mtA)) * NT + nt) * 32 + lane) * 8);
    const bf16x8 gp1 = *reinterpret_cast<const bf16x8*>(
        Gdp + ((((size_t)(t * 4 + mtA + 1)) * NT + nt) * 32 + lane) * 8);
#if HAVE_TDM
    if (wave == 0) tdm_wait();
#endif
    __syncthreads();

    f32x8 acc0 = {}, acc1 = {};
    for (int k = 0; k < H_SZ; k += 32) {
      bf16x16 bfrag = load_b_frag(lds_w, H_SZ, g * 16, k, lane);
      bf16x16 a0 = load_a_frag(lds_h, H_SZ, mtA * 16,      k, lane);
      bf16x16 a1 = load_a_frag(lds_h, H_SZ, mtA * 16 + 16, k, lane);
      acc0 = wmma_bf16(a0, bfrag, acc0);
      acc1 = wmma_bf16(a1, bfrag, acc1);
    }

    // gates = h@Whh^T (acc) + x-projection (gp) + bias -> LDS
    const int mh = (lane >> 4) << 3;
#pragma unroll
    for (int r = 0; r < 8; ++r) {
      const int m0 = mtA * 16 + r + mh;
      gbuf[g][m0][nc]      = acc0[r] + (float)gp0[r] + bias_n;
      gbuf[g][m0 + 16][nc] = acc1[r] + (float)gp1[r] + bias_n;
    }
    __syncthreads();

    // activations: 1024 h-elements owned by this block, 4 per thread
#pragma unroll
    for (int q = 0; q < 4; ++q) {
      const int e = tid + 256 * q;
      const int m = e >> 4, j = e & 15;
      float ig = gbuf[0][m][j], fg = gbuf[1][m][j];
      float gg = gbuf[2][m][j], og = gbuf[3][m][j];
      ig = 1.0f / (1.0f + __expf(-ig));
      fg = 1.0f / (1.0f + __expf(-fg));
      gg = tanhf(gg);
      og = 1.0f / (1.0f + __expf(-og));
      const float c = fg * cbuf[e] + ig * gg;
      cbuf[e] = c;
      const float h = og * tanhf(c);
      out[((size_t)t * B_SZ + m) * (2 * H_SZ) + dir * H_SZ + blk * 16 + j] = h;
      hout[(size_t)m * H_SZ + blk * 16 + j] = (bf16_t)h;
    }

    // device-scope barrier among this direction's 32 blocks
    __threadfence();
    __syncthreads();
    if (tid == 0) {
      __hip_atomic_fetch_add(&mycnt[s], 1u, __ATOMIC_RELEASE,
                             __HIP_MEMORY_SCOPE_AGENT);
      while (__hip_atomic_load(&mycnt[s], __ATOMIC_ACQUIRE,
                               __HIP_MEMORY_SCOPE_AGENT) <
             (uint32_t)SCAN_BLOCKS_PER_DIR)
        __builtin_amdgcn_s_sleep(1);
    }
    __syncthreads();
  }
}

// ---- Host launcher ---------------------------------------------------------

extern "C" void kernel_launch(void* const* d_in, const int* in_sizes, int n_in,
                              void* d_out, int out_size, void* d_ws,
                              size_t ws_size, hipStream_t stream) {
  (void)in_sizes; (void)n_in; (void)out_size;
  const float* x    = (const float*)d_in[0];
  const float* Wihf = (const float*)d_in[1];
  const float* Whhf = (const float*)d_in[2];
  const float* bihf = (const float*)d_in[3];
  const float* bhhf = (const float*)d_in[4];
  const float* Wihb = (const float*)d_in[5];
  const float* Whhb = (const float*)d_in[6];
  const float* bihb = (const float*)d_in[7];
  const float* bhhb = (const float*)d_in[8];
  float* out = (float*)d_out;

  char* ws = (char*)d_ws;
  size_t off = 0;
  auto carve = [&](size_t bytes) -> char* {
    char* p = ws + off;
    off = (off + bytes + 255) & ~(size_t)255;
    return p;
  };
  bf16_t*   Xb    = (bf16_t*)  carve((size_t)ROWS * I_SZ * 2);        //  32 MB
  bf16_t*   Wihfb = (bf16_t*)  carve((size_t)G4H * I_SZ * 2);         //   2 MB
  bf16_t*   Whhfb = (bf16_t*)  carve((size_t)G4H * H_SZ * 2);
  bf16_t*   Wihbb = (bf16_t*)  carve((size_t)G4H * I_SZ * 2);
  bf16_t*   Whhbb = (bf16_t*)  carve((size_t)G4H * H_SZ * 2);
  float*    biasw = (float*)   carve((size_t)2 * G4H * 4);
  bf16_t*   hbuf  = (bf16_t*)  carve((size_t)2 * 2 * B_SZ * H_SZ * 2);
  uint32_t* cnt   = (uint32_t*)carve((size_t)2 * T_LEN * 4);
  bf16_t*   G     = (bf16_t*)  carve((size_t)2 * ROWS * G4H * 2);     // 256 MB
  if (off > ws_size) return;  // workspace too small; nothing safe to do

  {
    const int nx = ROWS * I_SZ;                        // 16,777,216
    f32_to_bf16_kernel<<<(nx + 255) / 256, 256, 0, stream>>>(x, Xb, nx);
    const int nw = G4H * I_SZ;                         // 1,048,576
    f32_to_bf16_kernel<<<(nw + 255) / 256, 256, 0, stream>>>(Wihf, Wihfb, nw);
    f32_to_bf16_kernel<<<(nw + 255) / 256, 256, 0, stream>>>(Whhf, Whhfb, nw);
    f32_to_bf16_kernel<<<(nw + 255) / 256, 256, 0, stream>>>(Wihb, Wihbb, nw);
    f32_to_bf16_kernel<<<(nw + 255) / 256, 256, 0, stream>>>(Whhb, Whhbb, nw);
    bias_kernel<<<(2 * G4H + 255) / 256, 256, 0, stream>>>(bihf, bhhf, bihb,
                                                           bhhb, biasw);
    init_state_kernel<<<(2 * 2 * B_SZ * H_SZ) / 256, 256, 0, stream>>>(cnt, hbuf);
  }

  // phase 1: all-timestep input projections (chip-wide bf16 WMMA GEMM)
  input_gemm_kernel<<<2 * (ROWS / 64) * (G4H / 128), 256, 0, stream>>>(
      Xb, Wihfb, Wihbb, G);

  // phase 2: two independent recurrent scans, 32 blocks each
  lstm_scan_kernel<<<2 * SCAN_BLOCKS_PER_DIR, 256, 0, stream>>>(
      G, Whhfb, Whhbb, biasw, hbuf, cnt, out);
}
